// SlotMemoryCompressor_60352880443931
// MI455X (gfx1250) — matrix operand order repaired and checked by reading, not compiled, not run on hardware
//
#include <hip/hip_runtime.h>
#include <hip/hip_bf16.h>

#define HIDN 4096
#define SDIM 256
#define NSLOT 64
#define NB 4
#define TSEQ 4096
#define EPSILON 1e-5f
#define ATT_SCALE 0.0625f
#define NCHUNK 8
#define TCHUNK 512

typedef __attribute__((ext_vector_type(16))) __bf16 v16bf;
typedef __attribute__((ext_vector_type(8)))  __bf16 v8bf;
typedef __attribute__((ext_vector_type(8)))  float  v8f;

// ---------------------------------------------------------------- WMMA helpers

__device__ __forceinline__ v8f wmma_bf16(v16bf a, v16bf b, v8f c) {
  return __builtin_amdgcn_wmma_f32_16x16x32_bf16(false, a, false, b, (short)0, c,
                                                 false, false);
}

// A-matrix 16x32 bf16 fragment. Contiguous k order used here: lane half h=lane>>4, m=lane&15,
// element e: K = h*16 + e.  p points at &A[m*ld + k0 + h*16].
__device__ __forceinline__ v16bf afrag_f32(const float* p) {
  v16bf r;
#pragma unroll
  for (int w = 0; w < 8; ++w) { r[w] = (__bf16)p[w]; r[8 + w] = (__bf16)p[8 + w]; }
  return r;
}
__device__ __forceinline__ v16bf afrag_lds(const __bf16* p) {
  v8bf lo = *(const v8bf*)p;
  v8bf hi = *(const v8bf*)(p + 8);
  v16bf r;
#pragma unroll
  for (int i = 0; i < 8; ++i) { r[i] = lo[i]; r[8 + i] = hi[i]; }
  return r;
}
// B-matrix 32x16 bf16 fragment: n=lane&15, element e: K = h*16 + e.
// LDS layout [n][k]; p points at &B[n*ld + k0 + h*16].
__device__ __forceinline__ v16bf bfrag_f32(const float* p) {
  v16bf r;
#pragma unroll
  for (int e = 0; e < 16; ++e) r[e] = (__bf16)p[e];
  return r;
}
__device__ __forceinline__ v16bf bfrag_lds(const __bf16* p) {
  v8bf lo = *(const v8bf*)p;
  v8bf hi = *(const v8bf*)(p + 8);
  v16bf r;
#pragma unroll
  for (int i = 0; i < 8; ++i) { r[i] = lo[i]; r[8 + i] = hi[i]; }
  return r;
}
// B fragment from row-major [k][n] f32 source: p = &B[(k0+h*16)*ld + n].
__device__ __forceinline__ v16bf bfrag_strided_f32(const float* p, int ld) {
  v16bf r;
#pragma unroll
  for (int e = 0; e < 16; ++e) r[e] = (__bf16)p[(long)e * ld];
  return r;
}

// CDNA5 async global->LDS copy (16B per lane), tracked by ASYNCcnt.
__device__ __forceinline__ void async_copy_b128(const __bf16* gaddr, unsigned ldsaddr) {
  asm volatile("global_load_async_to_lds_b128 %0, %1, off"
               :: "v"(ldsaddr), "v"(gaddr) : "memory");
}
__device__ __forceinline__ void async_wait0() {
  asm volatile("s_wait_asynccnt 0x0" ::: "memory");
}

// ---------------------------------------------------------------- weight prep
// Tiled transpose+convert: dst[n][k] (bf16, ld=K) = src[k][n] (f32, ld=N).

__global__ __launch_bounds__(256) void cvtT_kernel(const float* __restrict__ src,
                                                   __bf16* __restrict__ dst,
                                                   int K, int N) {
  __shared__ float tile[32][33];
  const int kb = blockIdx.x * 32, nb = blockIdx.y * 32;
  const int tx = threadIdx.x & 31, ty = threadIdx.x >> 5;  // 32 x 8
#pragma unroll
  for (int i = 0; i < 32; i += 8)
    tile[ty + i][tx] = src[(long)(kb + ty + i) * N + nb + tx];
  __syncthreads();
#pragma unroll
  for (int i = 0; i < 32; i += 8)
    dst[(long)(nb + ty + i) * K + kb + tx] = (__bf16)tile[tx][ty + i];
}

__global__ void prep_gate_kernel(const float* __restrict__ gate_bias,
                                 float* __restrict__ gate_ws) {
  if (threadIdx.x == 0) {
    float g = 0.f;
    for (int i = 0; i < NSLOT; ++i) g += gate_bias[i];
    g *= (1.f / NSLOT);
    gate_ws[0] = 1.f / (1.f + __expf(-g));
  }
}

// ---------------------------------------------------------------- generic GEMM
// C[z] = A[z] (f32, MxK) @ BwT (bf16, [N][K] pre-transposed) [+ Cadd[z]]
// tiles 128x128x32, 8 waves. B tile DMA'd to LDS via async copy (no transpose
// needed, layout [n][k] matches the WMMA B-fragment read pattern).

__global__ __launch_bounds__(256) void gemm_bf16_kernel(
    const float* __restrict__ A, int lda, long sA,
    const __bf16* __restrict__ BwT, int ldbT,
    float* __restrict__ C, int ldc, long sC,
    const float* __restrict__ Cadd, int ldadd, long sAdd,
    int M, int N, int K) {
  __shared__ __bf16 As[128 * 40];
  __shared__ __bf16 Bs[128 * 40];
  const int z = blockIdx.z;
  A += (long)z * sA;
  C += (long)z * sC;
  if (Cadd) Cadd += (long)z * sAdd;
  const int m0 = blockIdx.x * 128;
  const int n0 = blockIdx.y * 128;
  const int tid = threadIdx.x;
  const int lane = tid & 31, wid = tid >> 5;
  const int h = lane >> 4, l15 = lane & 15;
  const int wm = wid & 3, wn = wid >> 2;  // 4(m) x 2(n) waves; wave tile 32x64

  v8f acc[2][4];
#pragma unroll
  for (int i = 0; i < 2; ++i)
#pragma unroll
    for (int j = 0; j < 4; ++j) acc[i][j] = (v8f){};

  const int ar0 = tid >> 3;          // 0..31
  const int ac0 = (tid & 7) * 4;     // 0..28
  const int bn = tid >> 2;           // 0..63 (row pair base)
  const int bks = tid & 3;           // 16B segment within 64B row

  for (int k0 = 0; k0 < K; k0 += 32) {
    // B tile: async DMA 128 rows x 64B, layout-preserving [n][k]
#pragma unroll
    for (int i = 0; i < 2; ++i) {
      int n = bn + i * 64;
      async_copy_b128(BwT + (long)(n0 + n) * ldbT + k0 + bks * 8,
                      (unsigned)(size_t)(&Bs[n * 40 + bks * 8]));
    }
    // A tile -> LDS [m][k], f32->bf16 (conversion forces the VALU path)
#pragma unroll
    for (int i = 0; i < 4; ++i) {
      int r = ar0 + i * 32;
      int gr = m0 + r;
      if (gr > M - 1) gr = M - 1;
      const float* ap = A + (long)gr * lda + k0 + ac0;
      const float4 v = *(const float4*)ap;
      if (k0 + 32 < K) __builtin_prefetch(ap + 32, 0, 1);  // next K-tile
      __bf16* d = &As[r * 40 + ac0];
      d[0] = (__bf16)v.x; d[1] = (__bf16)v.y; d[2] = (__bf16)v.z; d[3] = (__bf16)v.w;
    }
    async_wait0();
    __syncthreads();
    v16bf af[2], bff[4];
#pragma unroll
    for (int i = 0; i < 2; ++i)
      af[i] = afrag_lds(&As[(wm * 32 + i * 16 + l15) * 40 + h * 16]);
#pragma unroll
    for (int j = 0; j < 4; ++j)
      bff[j] = bfrag_lds(&Bs[(wn * 64 + j * 16 + l15) * 40 + h * 16]);
#pragma unroll
    for (int i = 0; i < 2; ++i)
#pragma unroll
      for (int j = 0; j < 4; ++j) acc[i][j] = wmma_bf16(af[i], bff[j], acc[i][j]);
    __syncthreads();
  }
  // store (C layout: lane->col, vgpr->row)
#pragma unroll
  for (int i = 0; i < 2; ++i)
#pragma unroll
    for (int j = 0; j < 4; ++j)
#pragma unroll
      for (int r = 0; r < 8; ++r) {
        int row = m0 + wm * 32 + i * 16 + h * 8 + r;
        int col = n0 + wn * 64 + j * 16 + l15;
        if (row < M) {
          float v = acc[i][j][r];
          if (Cadd) v += Cadd[(long)row * ldadd + col];
          C[(long)row * ldc + col] = v;
        }
      }
}

// ------------------------------------------------------- attention 1 (partial)
// scores = exp(q k^T /16) over a T-chunk; accumulate P row sums (denominator)
// and P@V partials. Deterministic: each (b,chunk) block owns its partial slot.

__global__ __launch_bounds__(256) void attn1_partial_kernel(
    const float* __restrict__ qm,          // [64,256]
    const float* __restrict__ kvr,         // [B*T,768] (k:0-255, v:256-511)
    float* __restrict__ denom_part,        // [B][NCHUNK][64]
    float* __restrict__ out_part) {        // [B][NCHUNK][64][256]
  __shared__ __bf16 Ps[64 * 72];
  __shared__ float red[64 * 4];
  __shared__ float denom_acc[64];
  const int chunk = blockIdx.x, b = blockIdx.y;
  const int tid = threadIdx.x, lane = tid & 31, wid = tid >> 5;
  const int h = lane >> 4, l15 = lane & 15;
  const int wm = wid & 3, wn = wid >> 2;
  const int sm0 = wm * 16;
  if (tid < 64) denom_acc[tid] = 0.f;

  // q A-fragments resident in VGPRs (K=256 -> 8 ksteps)
  v16bf aq[8];
#pragma unroll
  for (int ks = 0; ks < 8; ++ks)
    aq[ks] = afrag_f32(qm + (sm0 + l15) * 256 + ks * 32 + h * 16);

  v8f vacc[8];
#pragma unroll
  for (int j = 0; j < 8; ++j) vacc[j] = (v8f){};

  const float* kv_b = kvr + (long)b * TSEQ * 768;
  __syncthreads();

  for (int it = 0; it < TCHUNK / 64; ++it) {
    const int tbase = chunk * TCHUNK + it * 64;
    // scores [64s x 64t]; this wave: s-tile sm0, t cols wn*32 + j*16
    v8f sacc[2];
    sacc[0] = (v8f){}; sacc[1] = (v8f){};
#pragma unroll
    for (int j = 0; j < 2; ++j) {
      const int tcol = tbase + wn * 32 + j * 16 + l15;  // this lane's B row (n=t)
      const float* krow = kv_b + (long)tcol * 768;
#pragma unroll
      for (int ks = 0; ks < 8; ++ks) {
        v16bf bf = bfrag_f32(krow + ks * 32 + h * 16);
        sacc[j] = wmma_bf16(aq[ks], bf, sacc[j]);
      }
    }
    // exp -> LDS P[s][t] (bf16, A layout for second matmul)
#pragma unroll
    for (int j = 0; j < 2; ++j)
#pragma unroll
      for (int r = 0; r < 8; ++r) {
        float e = __expf(sacc[j][r] * ATT_SCALE);
        int s = sm0 + h * 8 + r;
        int t = wn * 32 + j * 16 + l15;
        Ps[s * 72 + t] = (__bf16)e;
      }
    __syncthreads();
    // denominator partial per s
    {
      int s = tid & 63, part = tid >> 6;
      float sum = 0.f;
#pragma unroll
      for (int i = 0; i < 16; ++i) sum += (float)Ps[s * 72 + part * 16 + i];
      red[s * 4 + part] = sum;
    }
    __syncthreads();
    if (tid < 64)
      denom_acc[tid] += red[tid * 4] + red[tid * 4 + 1] + red[tid * 4 + 2] + red[tid * 4 + 3];
    // out += P @ V  (K = 64 t's -> 2 ksteps); B frags strided from global v
#pragma unroll
    for (int ks = 0; ks < 2; ++ks) {
      v16bf aP = afrag_lds(&Ps[(sm0 + l15) * 72 + ks * 32 + h * 16]);
#pragma unroll
      for (int j = 0; j < 8; ++j) {
        int d = wn * 128 + j * 16 + l15;
        const float* vp = kv_b + (long)(tbase + ks * 32 + h * 16) * 768 + 256 + d;
        v16bf bf = bfrag_strided_f32(vp, 768);
        vacc[j] = wmma_bf16(aP, bf, vacc[j]);
      }
    }
    __syncthreads();
  }
  long base = ((long)b * NCHUNK + chunk) * 64;
  if (tid < 64) denom_part[base + tid] = denom_acc[tid];
#pragma unroll
  for (int j = 0; j < 8; ++j)
#pragma unroll
    for (int r = 0; r < 8; ++r) {
      int s = sm0 + h * 8 + r;
      int d = wn * 128 + j * 16 + l15;
      out_part[(base + s) * 256 + d] = vacc[j][r];
    }
}

__global__ void attn1_combine_kernel(const float* __restrict__ denom_part,
                                     const float* __restrict__ out_part,
                                     float* __restrict__ out_sd) {
  int idx = blockIdx.x * 256 + threadIdx.x;  // over 4*64*256
  int d = idx & 255;
  int s = (idx >> 8) & 63;
  int b = idx >> 14;
  float den = 0.f, o = 0.f;
  for (int c = 0; c < NCHUNK; ++c) {
    long base = ((long)b * NCHUNK + c) * 64;
    den += denom_part[base + s];
    o += out_part[(base + s) * 256 + d];
  }
  out_sd[((long)b * 64 + s) * 256 + d] = o / den;
}

// ------------------------------------------------------------------ row LN
__global__ __launch_bounds__(256) void ln_rows_kernel(float* __restrict__ X,
                                                      const float* __restrict__ w,
                                                      const float* __restrict__ bias,
                                                      int ncols) {
  __shared__ float red[256];
  float* x = X + (long)blockIdx.x * ncols;
  float s = 0.f, s2 = 0.f;
  for (int c = threadIdx.x; c < ncols; c += 256) {
    float v = x[c];
    s += v; s2 += v * v;
  }
  red[threadIdx.x] = s;
  __syncthreads();
  for (int o = 128; o > 0; o >>= 1) {
    if (threadIdx.x < o) red[threadIdx.x] += red[threadIdx.x + o];
    __syncthreads();
  }
  float mean = red[0] / ncols;
  __syncthreads();
  red[threadIdx.x] = s2;
  __syncthreads();
  for (int o = 128; o > 0; o >>= 1) {
    if (threadIdx.x < o) red[threadIdx.x] += red[threadIdx.x + o];
    __syncthreads();
  }
  float var = red[0] / ncols - mean * mean;
  float isig = rsqrtf(var + EPSILON);
  for (int c = threadIdx.x; c < ncols; c += 256)
    x[c] = (x[c] - mean) * isig * w[c] + bias[c];
}

// ------------------------------------------------------- attention 2 (recon)
// per (b, 64-row t-tile): softmax over S=64 slots, rout_sd = softmax @ rv.

__global__ __launch_bounds__(256) void attn2_kernel(
    const float* __restrict__ kvr,      // rq at cols 512-767
    const float* __restrict__ rkv,      // [B*64,512] (rk:0-255, rv:256-511)
    float* __restrict__ rout_sd) {      // [B*T,256]
  __shared__ __bf16 Ps[64 * 72];
  __shared__ float red[64 * 4];
  __shared__ float sums[64];
  const int ttile = blockIdx.x, b = blockIdx.y;
  const int t0 = ttile * 64;
  const int tid = threadIdx.x, lane = tid & 31, wid = tid >> 5;
  const int h = lane >> 4, l15 = lane & 15;
  const int wm = wid & 3, wn = wid >> 2;
  const int tm0 = wm * 16;
  const float* rq_b = kvr + ((long)b * TSEQ + t0) * 768 + 512;
  const float* rkv_b = rkv + (long)b * 64 * 512;

  v16bf aq[8];
#pragma unroll
  for (int ks = 0; ks < 8; ++ks)
    aq[ks] = afrag_f32(rq_b + (long)(tm0 + l15) * 768 + ks * 32 + h * 16);

  // scores [64t x 64s]
  v8f sacc[2];
  sacc[0] = (v8f){}; sacc[1] = (v8f){};
#pragma unroll
  for (int j = 0; j < 2; ++j) {
    const int scol = wn * 32 + j * 16 + l15;
    const float* rkrow = rkv_b + (long)scol * 512;
#pragma unroll
    for (int ks = 0; ks < 8; ++ks) {
      v16bf bf = bfrag_f32(rkrow + ks * 32 + h * 16);
      sacc[j] = wmma_bf16(aq[ks], bf, sacc[j]);
    }
  }
#pragma unroll
  for (int j = 0; j < 2; ++j)
#pragma unroll
    for (int r = 0; r < 8; ++r) {
      float e = __expf(sacc[j][r] * ATT_SCALE);
      int t = tm0 + h * 8 + r;
      int s = wn * 32 + j * 16 + l15;
      Ps[t * 72 + s] = (__bf16)e;
    }
  __syncthreads();
  {
    int t = tid & 63, part = tid >> 6;
    float sum = 0.f;
#pragma unroll
    for (int i = 0; i < 16; ++i) sum += (float)Ps[t * 72 + part * 16 + i];
    red[t * 4 + part] = sum;
  }
  __syncthreads();
  if (tid < 64)
    sums[tid] = red[tid * 4] + red[tid * 4 + 1] + red[tid * 4 + 2] + red[tid * 4 + 3];
  __syncthreads();

  v8f vacc[8];
#pragma unroll
  for (int j = 0; j < 8; ++j) vacc[j] = (v8f){};
#pragma unroll
  for (int ks = 0; ks < 2; ++ks) {
    v16bf aP = afrag_lds(&Ps[(tm0 + l15) * 72 + ks * 32 + h * 16]);
#pragma unroll
    for (int j = 0; j < 8; ++j) {
      int d = wn * 128 + j * 16 + l15;
      const float* vp = rkv_b + (long)(ks * 32 + h * 16) * 512 + 256 + d;
      v16bf bf = bfrag_strided_f32(vp, 512);
      vacc[j] = wmma_bf16(aP, bf, vacc[j]);
    }
  }
#pragma unroll
  for (int j = 0; j < 8; ++j)
#pragma unroll
    for (int r = 0; r < 8; ++r) {
      int t = tm0 + h * 8 + r;
      int d = wn * 128 + j * 16 + l15;
      rout_sd[((long)b * TSEQ + t0 + t) * 256 + d] = vacc[j][r] / sums[t];
    }
}

// --------------------------------------------- reconstruction GEMM, two modes
// MODE 1: rout_sd@Wro, accumulate per-row sum/sumsq (LN stats), no store.
// MODE 2: recompute, apply LN + gate, accumulate MSE vs segment.
// WroT is pre-transposed [4096][256] so B chunks are async-DMA'd to LDS.

template <int MODE>
__global__ __launch_bounds__(256) void rstage_kernel(
    const float* __restrict__ rout_sd,   // [16384,256]
    const __bf16* __restrict__ WroT,     // [4096][256] (pre-transposed)
    float* __restrict__ stats,           // [16384][2]
    const float* __restrict__ segment,   // [16384,4096]
    const float* __restrict__ rw, const float* __restrict__ rb,
    const float* __restrict__ gate_ws,
    float* __restrict__ loss_part) {
  __shared__ __bf16 As[64 * 264];
  __shared__ __bf16 Bs[64 * 136];
  __shared__ float sred[256];
  const int m0 = blockIdx.x * 64;
  const int tid = threadIdx.x, lane = tid & 31, wid = tid >> 5;
  const int h = lane >> 4, l15 = lane & 15;
  const int wm = wid & 3, wn = wid >> 2;  // 4(m) x 2(n); wave tile 16x32

  for (int idx = tid; idx < 64 * 256; idx += 256) {
    int r = idx >> 8, c = idx & 255;
    As[r * 264 + c] = (__bf16)rout_sd[(long)(m0 + r) * 256 + c];
  }
  float gate = (MODE == 2) ? gate_ws[0] : 0.f;
  float mu[8], isig[8], wsum[8], wsq[8];
#pragma unroll
  for (int r = 0; r < 8; ++r) { wsum[r] = 0.f; wsq[r] = 0.f; }
  if (MODE == 2) {
#pragma unroll
    for (int r = 0; r < 8; ++r) {
      int row = m0 + wm * 16 + h * 8 + r;
      float sm = stats[row * 2], sq = stats[row * 2 + 1];
      float m_ = sm * (1.f / HIDN);
      float var = sq * (1.f / HIDN) - m_ * m_;
      mu[r] = m_;
      isig[r] = rsqrtf(var + EPSILON);
    }
  }
  float lacc = 0.f;
  __syncthreads();

  for (int nt = 0; nt < 64; ++nt) {
    const int n0 = nt * 64;
    v8f acc[2];
    acc[0] = (v8f){}; acc[1] = (v8f){};
    for (int kh = 0; kh < 2; ++kh) {
      __syncthreads();
      // B chunk: 64 rows (n) x 128 k-halves via async DMA, layout [n][k]
#pragma unroll
      for (int i = 0; i < 4; ++i) {
        int seg = tid + i * 256;
        int n = seg >> 4, ks = seg & 15;
        async_copy_b128(WroT + (long)(n0 + n) * 256 + kh * 128 + ks * 8,
                        (unsigned)(size_t)(&Bs[n * 136 + ks * 8]));
      }
      async_wait0();
      __syncthreads();
#pragma unroll
      for (int ks = 0; ks < 4; ++ks) {
        v16bf aF = afrag_lds(&As[(wm * 16 + l15) * 264 + kh * 128 + ks * 32 + h * 16]);
#pragma unroll
        for (int j = 0; j < 2; ++j) {
          v16bf bF = bfrag_lds(&Bs[(wn * 32 + j * 16 + l15) * 136 + ks * 32 + h * 16]);
          acc[j] = wmma_bf16(aF, bF, acc[j]);
        }
      }
    }
    if (MODE == 1) {
#pragma unroll
      for (int j = 0; j < 2; ++j)
#pragma unroll
        for (int r = 0; r < 8; ++r) {
          float v = acc[j][r];
          wsum[r] += v;
          wsq[r] += v * v;
        }
    } else {
#pragma unroll
      for (int j = 0; j < 2; ++j) {
        int col = n0 + wn * 32 + j * 16 + l15;
        float w_ = rw[col], b_ = rb[col];
#pragma unroll
        for (int r = 0; r < 8; ++r) {
          int row = m0 + wm * 16 + h * 8 + r;
          float v = acc[j][r];
          float ln = (v - mu[r]) * isig[r] * w_ + b_;
          float diff = gate * ln - segment[(long)row * HIDN + col];
          lacc += diff * diff;
        }
      }
    }
  }

  if (MODE == 1) {
#pragma unroll
    for (int r = 0; r < 8; ++r)
#pragma unroll
      for (int m = 1; m < 16; m <<= 1) {
        wsum[r] += __shfl_xor(wsum[r], m, 32);
        wsq[r] += __shfl_xor(wsq[r], m, 32);
      }
    if (l15 == 0) {
#pragma unroll
      for (int r = 0; r < 8; ++r) {
        int lrow = wm * 16 + h * 8 + r;
        sred[(lrow * 2 + wn) * 2 + 0] = wsum[r];
        sred[(lrow * 2 + wn) * 2 + 1] = wsq[r];
      }
    }
    __syncthreads();
    if (tid < 64) {
      float s_ = sred[(tid * 2 + 0) * 2 + 0] + sred[(tid * 2 + 1) * 2 + 0];
      float q_ = sred[(tid * 2 + 0) * 2 + 1] + sred[(tid * 2 + 1) * 2 + 1];
      stats[(m0 + tid) * 2] = s_;
      stats[(m0 + tid) * 2 + 1] = q_;
    }
  } else {
    sred[tid] = lacc;
    __syncthreads();
    for (int o = 128; o > 0; o >>= 1) {
      if (tid < o) sred[tid] += sred[tid + o];
      __syncthreads();
    }
    if (tid == 0) loss_part[blockIdx.x] = sred[0];
  }
}

__global__ void finalize_kernel(const float* __restrict__ loss_part, int nparts,
                                const float* __restrict__ gate_bias,
                                float* __restrict__ out_tail) {
  __shared__ float red[256];
  float s = 0.f;
  for (int i = threadIdx.x; i < nparts; i += 256) s += loss_part[i];
  red[threadIdx.x] = s;
  __syncthreads();
  for (int o = 128; o > 0; o >>= 1) {
    if (threadIdx.x < o) red[threadIdx.x] += red[threadIdx.x + o];
    __syncthreads();
  }
  if (threadIdx.x == 0) {
    out_tail[0] = red[0] / ((float)NB * TSEQ * HIDN);
    float g = 0.f;
    for (int i = 0; i < NSLOT; ++i) g += gate_bias[i];
    g *= (1.f / NSLOT);
    out_tail[1] = 1.f / (1.f + __expf(-g));
  }
}

// ------------------------------------------------------------------ launcher

extern "C" void kernel_launch(void* const* d_in, const int* in_sizes, int n_in,
                              void* d_out, int out_size, void* d_ws, size_t ws_size,
                              hipStream_t stream) {
  const float* segment  = (const float*)d_in[0];   // [4,4096,4096]
  const float* slot_emb = (const float*)d_in[1];   // [64,4096]
  const float* Wq       = (const float*)d_in[2];
  const float* Wk       = (const float*)d_in[3];
  const float* Wv       = (const float*)d_in[4];
  const float* Wo       = (const float*)d_in[5];
  const float* Wrq      = (const float*)d_in[6];
  const float* Wro      = (const float*)d_in[7];
  const float* gate_b   = (const float*)d_in[8];
  const float* sn_w     = (const float*)d_in[9];
  const float* sn_b     = (const float*)d_in[10];
  const float* rn_w     = (const float*)d_in[11];
  const float* rn_b     = (const float*)d_in[12];

  float* out = (float*)d_out;
  float* updated = out;                             // [4,64,4096]
  float* out_tail = out + (long)NB * NSLOT * HIDN;  // loss, gate

  // workspace carve-up (256B aligned)
  char* p = (char*)d_ws;
  auto take = [&](size_t bytes) {
    char* r = p;
    p += (bytes + 255) & ~(size_t)255;
    return r;
  };
  __bf16* wkvrT  = (__bf16*)take((size_t)768 * HIDN * 2);   // [768][4096]
  __bf16* wqT    = (__bf16*)take((size_t)256 * HIDN * 2);   // [256][4096]
  __bf16* woT    = (__bf16*)take((size_t)HIDN * 256 * 2);   // [4096][256]
  __bf16* wroT   = (__bf16*)take((size_t)HIDN * 256 * 2);   // [4096][256]
  float* qbuf    = (float*)take((size_t)64 * 256 * 4);
  float* kvr     = (float*)take((size_t)NB * TSEQ * 768 * 4);
  float* den_part= (float*)take((size_t)NB * NCHUNK * 64 * 4);
  float* outp    = (float*)take((size_t)NB * NCHUNK * 64 * 256 * 4);
  float* out_sd  = (float*)take((size_t)NB * 64 * 256 * 4);
  float* rkv     = (float*)take((size_t)NB * 64 * 512 * 4);
  float* rout_sd = (float*)take((size_t)NB * TSEQ * 256 * 4);
  float* stats   = (float*)take((size_t)NB * TSEQ * 2 * 4);
  float* loss_p  = (float*)take((size_t)256 * 4);
  float* gate_ws = (float*)take(256);
  (void)ws_size; (void)in_sizes; (void)n_in; (void)out_size;

  // 1) weight prep: transpose+convert to bf16 [N][K] layouts; gate scalar
  cvtT_kernel<<<dim3(HIDN / 32, 256 / 32, 1), 256, 0, stream>>>(Wk, wkvrT, HIDN, 256);
  cvtT_kernel<<<dim3(HIDN / 32, 256 / 32, 1), 256, 0, stream>>>(
      Wv, wkvrT + (size_t)256 * HIDN, HIDN, 256);
  cvtT_kernel<<<dim3(HIDN / 32, 256 / 32, 1), 256, 0, stream>>>(
      Wrq, wkvrT + (size_t)512 * HIDN, HIDN, 256);
  cvtT_kernel<<<dim3(HIDN / 32, 256 / 32, 1), 256, 0, stream>>>(Wq, wqT, HIDN, 256);
  cvtT_kernel<<<dim3(256 / 32, HIDN / 32, 1), 256, 0, stream>>>(Wo, woT, 256, HIDN);
  cvtT_kernel<<<dim3(256 / 32, HIDN / 32, 1), 256, 0, stream>>>(Wro, wroT, 256, HIDN);
  prep_gate_kernel<<<1, 64, 0, stream>>>(gate_b, gate_ws);

  // 2) q = slot_emb @ Wq   [64,256]
  gemm_bf16_kernel<<<dim3(1, 2, 1), 256, 0, stream>>>(
      slot_emb, HIDN, 0, wqT, HIDN, qbuf, 256, 0, nullptr, 0, 0, 64, 256, HIDN);
  // 3) kvr = segment @ [Wk|Wv|Wrq]   [16384,768]
  gemm_bf16_kernel<<<dim3(128, 6, 1), 256, 0, stream>>>(
      segment, HIDN, 0, wkvrT, HIDN, kvr, 768, 0, nullptr, 0, 0,
      NB * TSEQ, 768, HIDN);
  // 4) slot attention partials + combine
  attn1_partial_kernel<<<dim3(NCHUNK, NB, 1), 256, 0, stream>>>(qbuf, kvr, den_part, outp);
  attn1_combine_kernel<<<dim3(256, 1, 1), 256, 0, stream>>>(den_part, outp, out_sd);
  // 5) updated_slots(pre-LN) = slots + out_sd @ Wo   (per batch)
  gemm_bf16_kernel<<<dim3(1, 32, NB), 256, 0, stream>>>(
      out_sd, 256, (long)64 * 256, woT, 256, updated, HIDN, (long)64 * HIDN,
      slot_emb, HIDN, 0, 64, HIDN, 256);
  // 6) LayerNorm in place -> updated_slots (final output)
  ln_rows_kernel<<<dim3(NB * NSLOT, 1, 1), 256, 0, stream>>>(updated, sn_w, sn_b, HIDN);
  // 7) rkv = updated_slots @ [Wk|Wv]   [256,512]
  gemm_bf16_kernel<<<dim3(2, 4, 1), 256, 0, stream>>>(
      updated, HIDN, 0, wkvrT, HIDN, rkv, 512, 0, nullptr, 0, 0,
      NB * NSLOT, 512, HIDN);
  // 8) reconstruction attention -> rout_sd  [16384,256]
  attn2_kernel<<<dim3(TSEQ / 64, NB, 1), 256, 0, stream>>>(kvr, rkv, rout_sd);
  // 9) LN stats pass (rout_sd @ Wro, row sum/sumsq)
  rstage_kernel<1><<<dim3(256, 1, 1), 256, 0, stream>>>(
      rout_sd, wroT, stats, segment, rn_w, rn_b, gate_ws, loss_p);
  // 10) LN + gate + MSE loss pass
  rstage_kernel<2><<<dim3(256, 1, 1), 256, 0, stream>>>(
      rout_sd, wroT, stats, segment, rn_w, rn_b, gate_ws, loss_p);
  // 11) finalize loss + gate
  finalize_kernel<<<1, 256, 0, stream>>>(loss_p, 256, gate_b, out_tail);
}
